// QRNNLayer_60086592471416
// MI455X (gfx1250) — matrix operand order repaired and checked by reading, not compiled, not run on hardware
//
#include <hip/hip_runtime.h>

// ============================================================================
// QRNN layer for MI455X (gfx1250, wave32, WMMA).
//
//   1) split_bf16_kernel : X, W (f32) -> bf16 hi/lo pairs in d_ws (bf16x2
//      error-compensated split: ~16 effective mantissa bits).
//   2) qrnn_gemm_bf16x2  : Y = X*W^T + b via v_wmma_f32_16x16x32_bf16.
//      Wave tile 32x64 (8 f32 accumulators, 24 WMMA : 24 b128 loads per K=32
//      step). Block = 8 waves stacked in M over one 64-col strip, so all
//      waves share B fragments through WGP$ (L0) -> ~16x less L2 B-traffic
//      than a 16-row block layout. Fused epilogue:
//        tanh(Z)    -> out C region   (cols    0..1023)
//        sigmoid(F) -> ws F buffer    (cols 1024..2047)
//        sigmoid(O) -> out H region   (cols 2048..3071)
//   3) Blocked (work-efficient) scan for c_t = f*z + (1-f)*c_{t-1}:
//      the recurrence is affine (c = a*c + b), so compose 16 segments in
//      parallel (4096 waves -> memory latency hidden), prefix the 16 segment
//      carries per channel, then re-apply writing C and H = sigmoid(O)*C.
// ============================================================================

#define SEQ   4096
#define NB    8
#define NIN   1024
#define NHID  1024
#define NCOLS (3 * NHID)       // 3072
#define MROWS (SEQ * NB)       // 32768
#define NCH   (NB * NHID)      // 8192 independent scan channels
#define SEG   16               // scan segments
#define SLEN  (SEQ / SEG)      // 256 steps per segment

typedef __bf16 v4bf  __attribute__((ext_vector_type(4)));
typedef __bf16 v8bf  __attribute__((ext_vector_type(8)));
typedef __bf16 v16bf __attribute__((ext_vector_type(16)));
typedef float  v8f   __attribute__((ext_vector_type(8)));

__device__ __forceinline__ float fsigmoid(float x) {
  return 1.0f / (1.0f + __expf(-x));
}
__device__ __forceinline__ float ftanh_fast(float x) {
  float e = __expf(-2.0f * fabsf(x));     // e in (0, 1], no overflow
  float t = (1.0f - e) / (1.0f + e);
  return copysignf(t, x);
}

// ---------------------------------------------------------------------------
// Kernel 1: split f32 -> (bf16 hi, bf16 lo), 4 elements/thread.
// ---------------------------------------------------------------------------
__global__ __launch_bounds__(256) void split_bf16_kernel(
    const float* __restrict__ src, __bf16* __restrict__ hi,
    __bf16* __restrict__ lo, int n4) {
  int i = blockIdx.x * blockDim.x + threadIdx.x;
  if (i >= n4) return;
  float4 v = reinterpret_cast<const float4*>(src)[i];
  float f[4] = {v.x, v.y, v.z, v.w};
  v4bf h, l;
#pragma unroll
  for (int j = 0; j < 4; ++j) {
    __bf16 hh = (__bf16)f[j];
    h[j] = hh;
    l[j] = (__bf16)(f[j] - (float)hh);
  }
  reinterpret_cast<v4bf*>(hi)[i] = h;
  reinterpret_cast<v4bf*>(lo)[i] = l;
}

// ---------------------------------------------------------------------------
// Fragment loaders matching CDNA5 16-bit WMMA VGPR layouts (wave32).
// A (16x32 MxK): lane<16 -> K = {k0+0..7, k0+16..23}; lanes>=16 offset by +8
//   (caller pre-offsets pointer by half*8; second chunk at +16 elements).
// B (32x16 KxN): same K split as A (ld16_gap, pointer pre-offset by half*8).
// ---------------------------------------------------------------------------
__device__ __forceinline__ v16bf ld16_gap(const __bf16* p) {
  v8bf a = *reinterpret_cast<const v8bf*>(p);
  v8bf b = *reinterpret_cast<const v8bf*>(p + 16);
  v16bf r;
#pragma unroll
  for (int i = 0; i < 8; ++i) { r[i] = a[i]; r[i + 8] = b[i]; }
  return r;
}
__device__ __forceinline__ v16bf ld16_contig(const __bf16* p) {
  v8bf a = *reinterpret_cast<const v8bf*>(p);
  v8bf b = *reinterpret_cast<const v8bf*>(p + 8);
  v16bf r;
#pragma unroll
  for (int i = 0; i < 8; ++i) { r[i] = a[i]; r[i + 8] = b[i]; }
  return r;
}

__device__ __forceinline__ v8f wmma_bf16(v16bf a, v16bf b, v8f c) {
  return __builtin_amdgcn_wmma_f32_16x16x32_bf16(
      /*neg_a=*/false, a, /*neg_b=*/false, b,
      /*c_mod=*/(short)0, c, /*reuse_a=*/false, /*reuse_b=*/false);
}

// ---------------------------------------------------------------------------
// Kernel 2: GEMM + bias + activations.
// Block = 256 threads = 8 waves, block tile 256(M) x 64(N), wave tile 32x64.
// Per K=32 step per wave: 6 WMMA x 4 N-subtiles = 24 v_wmma, 24 b128 loads.
// ---------------------------------------------------------------------------
__global__ __launch_bounds__(256) void qrnn_gemm_bf16x2(
    const __bf16* __restrict__ Xhi, const __bf16* __restrict__ Xlo,
    const __bf16* __restrict__ Whi, const __bf16* __restrict__ Wlo,
    const float* __restrict__ bias,
    float* __restrict__ Hbuf,    // sigmoid(O)
    float* __restrict__ Cbuf,    // tanh(Z)
    float* __restrict__ Fbuf) {  // sigmoid(F)
  const int lane = threadIdx.x & 31;
  const int wave = threadIdx.x >> 5;
  const int lm   = lane & 15;
  const int half = lane >> 4;

  const int m0 = blockIdx.y * 256 + wave * 32;   // waves stacked in M
  const int n0 = blockIdx.x * 64;                // block-shared N strip

  const __bf16* pAh[2];
  const __bf16* pAl[2];
#pragma unroll
  for (int mi = 0; mi < 2; ++mi) {
    size_t arow = (size_t)(m0 + mi * 16 + lm) * NIN;
    pAh[mi] = Xhi + arow + half * 8;
    pAl[mi] = Xlo + arow + half * 8;
  }
  const __bf16* pBh[4];
  const __bf16* pBl[4];
#pragma unroll
  for (int j = 0; j < 4; ++j) {
    size_t brow = (size_t)(n0 + j * 16 + lm) * NIN;
    pBh[j] = Whi + brow + half * 8;
    pBl[j] = Wlo + brow + half * 8;
  }

  v8f acc[2][4] = {{{}, {}, {}, {}}, {{}, {}, {}, {}}};

#pragma unroll 2
  for (int k0 = 0; k0 < NIN; k0 += 32) {
    v16bf ah0 = ld16_gap(pAh[0] + k0);
    v16bf al0 = ld16_gap(pAl[0] + k0);
    v16bf ah1 = ld16_gap(pAh[1] + k0);
    v16bf al1 = ld16_gap(pAl[1] + k0);
#pragma unroll
    for (int j = 0; j < 4; ++j) {
      v16bf bh = ld16_gap(pBh[j] + k0);
      v16bf bl = ld16_gap(pBl[j] + k0);
      acc[0][j] = wmma_bf16(ah0, bh, acc[0][j]);   // hi*hi
      acc[1][j] = wmma_bf16(ah1, bh, acc[1][j]);
      acc[0][j] = wmma_bf16(ah0, bl, acc[0][j]);   // hi*lo correction
      acc[1][j] = wmma_bf16(ah1, bl, acc[1][j]);
      acc[0][j] = wmma_bf16(al0, bh, acc[0][j]);   // lo*hi correction
      acc[1][j] = wmma_bf16(al1, bh, acc[1][j]);
    }
  }

  // Epilogue. 64-col wave strip (64 | 1024) lies entirely in one region.
  const int region = n0 >> 10;  // 0 = Z, 1 = F, 2 = O
  float* dst = (region == 0) ? Cbuf : ((region == 1) ? Fbuf : Hbuf);

#pragma unroll
  for (int mi = 0; mi < 2; ++mi) {
#pragma unroll
    for (int j = 0; j < 4; ++j) {
      const int col = n0 + j * 16 + lm;
      const float bv = bias[col];
      const int h = col & (NHID - 1);
#pragma unroll
      for (int v = 0; v < 8; ++v) {
        const int row = m0 + mi * 16 + v + 8 * half;  // C/D: VGPR v -> M=v(+8)
        const float y = acc[mi][j][v] + bv;
        const float r = (region == 0) ? ftanh_fast(y) : fsigmoid(y);
        dst[(size_t)row * NHID + h] = r;
      }
    }
  }
}

// ---------------------------------------------------------------------------
// Scan phase A: per (segment, channel), compose the affine map of the
// segment: c_out = A*c_in + B with per-step map c' = (1-f)*c + f*z.
// 131072 threads -> 4096 waves: load latency fully hidden by occupancy.
// ---------------------------------------------------------------------------
__global__ __launch_bounds__(256) void scan_compose(
    const float* __restrict__ Fbuf, const float* __restrict__ Zbuf,
    float* __restrict__ Aseg, float* __restrict__ Bseg) {
  const int g = blockIdx.x * blockDim.x + threadIdx.x;  // g = p*NCH + t
  if (g >= SEG * NCH) return;
  const int p = g >> 13;           // / NCH
  const int t = g & (NCH - 1);
  size_t idx = (size_t)p * SLEN * NCH + t;
  float A = 1.0f, B = 0.0f;
  for (int s = 0; s < SLEN; ++s) {
    const float f = Fbuf[idx];
    const float z = Zbuf[idx];
    const float a2 = 1.0f - f;
    A = a2 * A;                    // compose (a2, f*z) after (A, B)
    B = fmaf(a2, B, f * z);
    idx += NCH;
  }
  Aseg[g] = A;
  Bseg[g] = B;
}

// ---------------------------------------------------------------------------
// Scan phase B: per channel, prefix over the 16 segment maps -> carry-in per
// segment, plus the final state C[-1].
// ---------------------------------------------------------------------------
__global__ __launch_bounds__(256) void scan_carry(
    const float* __restrict__ hidden, const float* __restrict__ Aseg,
    const float* __restrict__ Bseg, float* __restrict__ carry,
    float* __restrict__ Clast) {
  const int t = blockIdx.x * blockDim.x + threadIdx.x;
  if (t >= NCH) return;
  float c = hidden[t];
#pragma unroll
  for (int p = 0; p < SEG; ++p) {
    carry[p * NCH + t] = c;        // carry-in of segment p
    c = fmaf(Aseg[p * NCH + t], c, Bseg[p * NCH + t]);
  }
  Clast[t] = c;
}

// ---------------------------------------------------------------------------
// Scan phase C: re-apply within segments, overwriting z->C and sig(O)->H.
// ---------------------------------------------------------------------------
__global__ __launch_bounds__(256) void scan_apply(
    const float* __restrict__ Fbuf, float* __restrict__ Cbuf,
    float* __restrict__ Hbuf, const float* __restrict__ carry) {
  const int g = blockIdx.x * blockDim.x + threadIdx.x;
  if (g >= SEG * NCH) return;
  const int p = g >> 13;
  const int t = g & (NCH - 1);
  float c = carry[g];
  size_t idx = (size_t)p * SLEN * NCH + t;
  for (int s = 0; s < SLEN; ++s) {
    const float z = Cbuf[idx];
    const float f = Fbuf[idx];
    const float o = Hbuf[idx];
    c = fmaf(f, z - c, c);         // f*z + (1-f)*c
    Cbuf[idx] = c;
    Hbuf[idx] = o * c;
    idx += NCH;
  }
}

// ---------------------------------------------------------------------------
extern "C" void kernel_launch(void* const* d_in, const int* in_sizes, int n_in,
                              void* d_out, int out_size, void* d_ws,
                              size_t ws_size, hipStream_t stream) {
  const float* X      = (const float*)d_in[0];  // [SEQ, NB, NIN]
  const float* hidden = (const float*)d_in[1];  // [NB, NHID]
  const float* W      = (const float*)d_in[2];  // [3*NHID, NIN]
  const float* bias   = (const float*)d_in[3];  // [3*NHID]

  const size_t SBH = (size_t)SEQ * NB * NHID;   // 33,554,432
  const size_t SBI = (size_t)SEQ * NB * NIN;
  const size_t WN  = (size_t)NCOLS * NIN;

  float* out   = (float*)d_out;
  float* Hbuf  = out;                           // [SEQ, NB, NHID]
  float* Clast = out + SBH;                     // [1, NB, NHID]
  float* Cbuf  = out + SBH + (size_t)NCH;       // [SEQ, NB, NHID]

  // Workspace: Xhi | Xlo | Whi | Wlo | Fbuf | Aseg | Bseg | carry
  __bf16* Xhi  = (__bf16*)d_ws;
  __bf16* Xlo  = Xhi + SBI;
  __bf16* Whi  = Xlo + SBI;
  __bf16* Wlo  = Whi + WN;
  float*  Fbuf = (float*)(Wlo + WN);
  float*  Aseg = Fbuf + SBH;
  float*  Bseg = Aseg + (size_t)SEG * NCH;
  float*  carry = Bseg + (size_t)SEG * NCH;

  // 1) Split conversions (memory-bound, ~290 MB of traffic, ~12 us).
  {
    int n4 = (int)(SBI / 4);
    split_bf16_kernel<<<(n4 + 255) / 256, 256, 0, stream>>>(X, Xhi, Xlo, n4);
  }
  {
    int n4 = (int)(WN / 4);
    split_bf16_kernel<<<(n4 + 255) / 256, 256, 0, stream>>>(W, Whi, Wlo, n4);
  }

  // 2) GEMM + activations: grid (3072/64, 32768/256) = (48, 128),
  //    N fastest so the 48 blocks of an M-stripe reuse A through L2.
  {
    dim3 grid(NCOLS / 64, MROWS / 256);
    qrnn_gemm_bf16x2<<<grid, 256, 0, stream>>>(Xhi, Xlo, Whi, Wlo, bias,
                                               Hbuf, Cbuf, Fbuf);
  }

  // 3) Blocked scan.
  {
    int n = SEG * NCH;  // 131072
    scan_compose<<<n / 256, 256, 0, stream>>>(Fbuf, Cbuf, Aseg, Bseg);
    scan_carry<<<NCH / 256, 256, 0, stream>>>(hidden, Aseg, Bseg, carry,
                                              Clast);
    scan_apply<<<n / 256, 256, 0, stream>>>(Fbuf, Cbuf, Hbuf, carry);
  }
}
